// CrossModalAttentionImproved_50749333570249
// MI455X (gfx1250) — compile-verified, matrix operand order checked by reading, not run on hardware
//
#include <hip/hip_runtime.h>
#include <hip/hip_bf16.h>

// ---------------------------------------------------------------------------
// MI455X (gfx1250): bf16 WMMA GEMMs (all NT: B pre-transposed to [N,K]) with
// async global->LDS staging when available, + elementwise epilogues.
// D = 768, H = 3072, B = 16, N = 2048, M = B*N = 32768.
// ---------------------------------------------------------------------------

#ifndef __has_builtin
#define __has_builtin(x) 0
#endif
#if __has_builtin(__builtin_amdgcn_global_load_async_to_lds_b128) && \
    __has_builtin(__builtin_amdgcn_s_wait_asynccnt)
#define USE_ASYNC_LDS 1
#else
#define USE_ASYNC_LDS 0
#endif

typedef __attribute__((ext_vector_type(16))) __bf16 bf16x16;
typedef __attribute__((ext_vector_type(8)))  __bf16 bf16x8;
typedef __attribute__((ext_vector_type(4)))  __bf16 bf16x4;
typedef __attribute__((ext_vector_type(8)))  float  floatx8;
typedef __attribute__((ext_vector_type(4)))  float  floatx4;
typedef __attribute__((ext_vector_type(4)))  int    intx4;

// -------------------------- f32 -> bf16 conversion -------------------------
__global__ __launch_bounds__(256) void cvt_f32_bf16(const float* __restrict__ x,
                                                    __bf16* __restrict__ y,
                                                    long long n4 /* n/4 */) {
    long long i = (long long)blockIdx.x * 256 + threadIdx.x;
    if (i >= n4) return;
    floatx4 v = *(const floatx4*)(x + 4 * i);
    bf16x4 o;
    o[0] = (__bf16)v[0]; o[1] = (__bf16)v[1];
    o[2] = (__bf16)v[2]; o[3] = (__bf16)v[3];
    *(bf16x4*)(y + 4 * i) = o;
}

// ------------------ transpose (f32 or bf16 src) -> bf16 --------------------
// src [R,C] -> dst [C,R]; 32x32 LDS tiles; batched via blockIdx.z.
template <typename SrcT>
__global__ __launch_bounds__(256) void transpose_to_bf16(
    const SrcT* __restrict__ src, __bf16* __restrict__ dst,
    int R, int C, long long sS, long long sD)
{
    __shared__ float t[32][33];
    src += (long long)blockIdx.z * sS;
    dst += (long long)blockIdx.z * sD;
    const int c0 = blockIdx.x * 32, r0 = blockIdx.y * 32;
    const int tx = threadIdx.x & 31, ty = threadIdx.x >> 5;
    #pragma unroll
    for (int i = 0; i < 4; ++i) {
        const int r = ty + i * 8;
        t[r][tx] = (float)src[(long long)(r0 + r) * C + c0 + tx];
    }
    __syncthreads();
    #pragma unroll
    for (int i = 0; i < 4; ++i) {
        const int r = ty + i * 8;
        dst[(long long)(c0 + r) * R + r0 + tx] = (__bf16)t[tx][r];
    }
}

// ------------------------------ WMMA GEMM (NT) -----------------------------
// C[M,N] = alpha * A[M,K] * Bt[N,K]^T + bias. A,Bt bf16 row-major; C f32 or
// bf16. Batched via blockIdx.z with element strides sA/sB/sC.
static constexpr int TBM = 128;
static constexpr int TBN = 128;
static constexpr int TBK = 64;    // two WMMA K-steps per barrier round
static constexpr int LSTR = 72;   // padded LDS row stride (144B): conflict-free b128

template <bool OUTBF>
__global__ __launch_bounds__(256) void gemm_nt_wmma(
    const __bf16* __restrict__ A, const __bf16* __restrict__ Bt,
    const float* __restrict__ bias,
    float* __restrict__ Cf, __bf16* __restrict__ Cb,
    int M, int N, int K, float alpha,
    long long sA, long long sB, long long sC)
{
    __shared__ __bf16 As[TBM * LSTR];
    __shared__ __bf16 Bs[TBN * LSTR];

    const int tid  = threadIdx.x;
    const int lane = tid & 31;
    const int wid  = tid >> 5;
    const int bz   = blockIdx.z;
    A  += (long long)bz * sA;
    Bt += (long long)bz * sB;

    const int m_blk = blockIdx.y * TBM;
    const int n_blk = blockIdx.x * TBN;

    // 8 waves arranged 2(M) x 4(N); each wave owns a 64x32 output sub-tile.
    const int wm = (wid >> 2) * 64;
    const int wn = (wid & 3) * 32;

    const int lh  = lane >> 4;   // lane half selects K 8..15 / 24..31 rows
    const int r16 = lane & 15;

    floatx8 c[4][2] = {};

    for (int kb = 0; kb < K; kb += TBK) {
        // ---- stage A and Bt tiles: 128 rows x 64 bf16 each.
        //      1024 16-byte chunks per tile; 256 threads x 4 chunks. ----
        #pragma unroll
        for (int it = 0; it < 4; ++it) {
            const int chunk = tid + it * 256;
            const int row   = chunk >> 3;
            const int cc    = (chunk & 7) * 8;
            const __bf16* ga = A  + (long long)(m_blk + row) * K + kb + cc;
            const __bf16* gb = Bt + (long long)(n_blk + row) * K + kb + cc;
            __bf16* la = &As[row * LSTR + cc];
            __bf16* lb = &Bs[row * LSTR + cc];
#if USE_ASYNC_LDS
            // signature (from compiler diagnostic): (int4*, int4*, imm, imm)
            __builtin_amdgcn_global_load_async_to_lds_b128(
                (intx4*)ga, (intx4*)la, 0, 0);
            __builtin_amdgcn_global_load_async_to_lds_b128(
                (intx4*)gb, (intx4*)lb, 0, 0);
#else
            *(bf16x8*)la = *(const bf16x8*)ga;
            *(bf16x8*)lb = *(const bf16x8*)gb;
            __builtin_prefetch(ga + TBK, 0, 1);   // global_prefetch_b8 next K tile
            __builtin_prefetch(gb + TBK, 0, 1);
#endif
        }
#if USE_ASYNC_LDS
        __builtin_amdgcn_s_wait_asynccnt(0);
#endif
        __syncthreads();

        // ---- 2 WMMA K-steps; ISA 16-bit A/B layout: lanes<16 hold K0..7 &
        //      K16..23 of their row; lanes>=16 hold K8..15 & K24..31. ----
        #pragma unroll
        for (int ks = 0; ks < TBK; ks += 32) {
            bf16x16 a[4], b[2];
            #pragma unroll
            for (int i = 0; i < 4; ++i) {
                const __bf16* rp = &As[(wm + i * 16 + r16) * LSTR + ks + 8 * lh];
                union { bf16x16 v; bf16x8 h[2]; } u;
                u.h[0] = *(const bf16x8*)rp;
                u.h[1] = *(const bf16x8*)(rp + 16);
                a[i] = u.v;
            }
            #pragma unroll
            for (int j = 0; j < 2; ++j) {
                const __bf16* rp = &Bs[(wn + j * 16 + r16) * LSTR + ks + 8 * lh];
                union { bf16x16 v; bf16x8 h[2]; } u;
                u.h[0] = *(const bf16x8*)rp;
                u.h[1] = *(const bf16x8*)(rp + 16);
                b[j] = u.v;
            }
            #pragma unroll
            for (int i = 0; i < 4; ++i)
                #pragma unroll
                for (int j = 0; j < 2; ++j)
                    c[i][j] = __builtin_amdgcn_wmma_f32_16x16x32_bf16(
                        false, a[i], false, b[j], (short)0, c[i][j], false, false);
        }
        __syncthreads();
    }

    // ---- epilogue: C fragment (VGPR e: M = e + 8*laneHalf, N = lane&15) ----
    #pragma unroll
    for (int i = 0; i < 4; ++i) {
        #pragma unroll
        for (int j = 0; j < 2; ++j) {
            const int col = n_blk + wn + j * 16 + r16;
            const float bv = bias ? bias[col] : 0.0f;
            #pragma unroll
            for (int e = 0; e < 8; ++e) {
                const int row = m_blk + wm + i * 16 + e + 8 * lh;
                const float val = c[i][j][e] * alpha + bv;
                const long long idx = (long long)bz * sC + (long long)row * N + col;
                if (OUTBF) Cb[idx] = (__bf16)val;
                else       Cf[idx] = val;
            }
        }
    }
}

// ------------------------- row softmax: f32 -> bf16 ------------------------
__global__ __launch_bounds__(256) void softmax_rows_2048(
    const float* __restrict__ S, __bf16* __restrict__ P)
{
    const long long row = blockIdx.x;
    const float* sr = S + row * 2048;
    __bf16* pr = P + row * 2048;
    const int tid = threadIdx.x;
    __shared__ float red[256];

    float v[8];
    float mx = -3.4e38f;
    #pragma unroll
    for (int i = 0; i < 8; ++i) { v[i] = sr[tid + i * 256]; mx = fmaxf(mx, v[i]); }
    red[tid] = mx; __syncthreads();
    for (int o = 128; o > 0; o >>= 1) { if (tid < o) red[tid] = fmaxf(red[tid], red[tid + o]); __syncthreads(); }
    mx = red[0]; __syncthreads();

    float sum = 0.0f;
    #pragma unroll
    for (int i = 0; i < 8; ++i) { v[i] = __expf(v[i] - mx); sum += v[i]; }
    red[tid] = sum; __syncthreads();
    for (int o = 128; o > 0; o >>= 1) { if (tid < o) red[tid] += red[tid + o]; __syncthreads(); }
    const float inv = 1.0f / red[0];

    #pragma unroll
    for (int i = 0; i < 8; ++i) pr[tid + i * 256] = (__bf16)(v[i] * inv);
}

// --------------------- residual add + LayerNorm ----------------------------
template <int COLS>
__global__ __launch_bounds__(256) void add_layernorm(
    const float* __restrict__ R, const float* __restrict__ Y,
    const float* __restrict__ g, const float* __restrict__ b,
    float* __restrict__ outf, __bf16* __restrict__ outb)
{
    constexpr int E = COLS / 256;
    const long long row = blockIdx.x;
    const float* rr = R + row * COLS;
    const float* yr = Y + row * COLS;
    const int tid = threadIdx.x;
    __shared__ float red[256];

    float v[E];
    float s = 0.0f;
    #pragma unroll
    for (int i = 0; i < E; ++i) { v[i] = rr[tid + i * 256] + yr[tid + i * 256]; s += v[i]; }
    red[tid] = s; __syncthreads();
    for (int o = 128; o > 0; o >>= 1) { if (tid < o) red[tid] += red[tid + o]; __syncthreads(); }
    const float mean = red[0] * (1.0f / COLS); __syncthreads();

    float s2 = 0.0f;
    #pragma unroll
    for (int i = 0; i < E; ++i) { float d = v[i] - mean; s2 += d * d; }
    red[tid] = s2; __syncthreads();
    for (int o = 128; o > 0; o >>= 1) { if (tid < o) red[tid] += red[tid + o]; __syncthreads(); }
    const float inv = __frsqrt_rn(red[0] * (1.0f / COLS) + 1e-5f);

    #pragma unroll
    for (int i = 0; i < E; ++i) {
        const int cc = tid + i * 256;
        const float o = (v[i] - mean) * inv * g[cc] + b[cc];
        outf[row * COLS + cc] = o;
        if (outb) outb[row * COLS + cc] = (__bf16)o;
    }
}

// ---------------- LayerNorm + exact GELU -> bf16 ----------------------------
template <int COLS>
__global__ __launch_bounds__(256) void layernorm_gelu(
    const float* __restrict__ X, const float* __restrict__ g,
    const float* __restrict__ b, __bf16* __restrict__ out)
{
    constexpr int E = COLS / 256;
    const long long row = blockIdx.x;
    const float* xr = X + row * COLS;
    const int tid = threadIdx.x;
    __shared__ float red[256];

    float v[E];
    float s = 0.0f;
    #pragma unroll
    for (int i = 0; i < E; ++i) { v[i] = xr[tid + i * 256]; s += v[i]; }
    red[tid] = s; __syncthreads();
    for (int o = 128; o > 0; o >>= 1) { if (tid < o) red[tid] += red[tid + o]; __syncthreads(); }
    const float mean = red[0] * (1.0f / COLS); __syncthreads();

    float s2 = 0.0f;
    #pragma unroll
    for (int i = 0; i < E; ++i) { float d = v[i] - mean; s2 += d * d; }
    red[tid] = s2; __syncthreads();
    for (int o = 128; o > 0; o >>= 1) { if (tid < o) red[tid] += red[tid + o]; __syncthreads(); }
    const float inv = __frsqrt_rn(red[0] * (1.0f / COLS) + 1e-5f);

    #pragma unroll
    for (int i = 0; i < E; ++i) {
        const int cc = tid + i * 256;
        const float x = (v[i] - mean) * inv * g[cc] + b[cc];
        const float ge = 0.5f * x * (1.0f + erff(x * 0.70710678118654752f)); // exact GELU
        out[row * COLS + cc] = (__bf16)ge;
    }
}

// ------------------------------- launcher ----------------------------------
extern "C" void kernel_launch(void* const* d_in, const int* in_sizes, int n_in,
                              void* d_out, int out_size, void* d_ws, size_t ws_size,
                              hipStream_t stream) {
    constexpr int B = 16, N = 2048, D = 768, H = 3072;
    constexpr int M = B * N;                       // 32768 rows
    const long long nBND = (long long)M * D;
    const long long nBNH = (long long)M * H;
    const long long nS   = (long long)B * N * N;

    const float* q_in = (const float*)d_in[0];
    const float* k_in = (const float*)d_in[1];
    const float* Wq = (const float*)d_in[2];  const float* bq = (const float*)d_in[3];
    const float* Wk = (const float*)d_in[4];  const float* bk = (const float*)d_in[5];
    const float* Wv = (const float*)d_in[6];  const float* bv = (const float*)d_in[7];
    const float* Wo = (const float*)d_in[8];  const float* bo = (const float*)d_in[9];
    const float* g1 = (const float*)d_in[10]; const float* b1 = (const float*)d_in[11];
    const float* W1 = (const float*)d_in[12]; const float* bf1 = (const float*)d_in[13];
    const float* gf = (const float*)d_in[14]; const float* bff = (const float*)d_in[15];
    const float* W2 = (const float*)d_in[16]; const float* bf2 = (const float*)d_in[17];
    const float* g2 = (const float*)d_in[18]; const float* b2 = (const float*)d_in[19];

    // ---- carve workspace ----
    char* wp = (char*)d_ws;
    auto carve = [&](long long bytes) -> void* {
        void* p = (void*)wp;
        wp += (bytes + 255) & ~255LL;
        return p;
    };
    __bf16* Xq  = (__bf16*)carve(nBND * 2);
    __bf16* Xk  = (__bf16*)carve(nBND * 2);
    __bf16* WqT = (__bf16*)carve((long long)D * D * 2);   // [D,D]^T
    __bf16* WkT = (__bf16*)carve((long long)D * D * 2);
    __bf16* WvT = (__bf16*)carve((long long)D * D * 2);
    __bf16* WoT = (__bf16*)carve((long long)D * D * 2);
    __bf16* W1T = (__bf16*)carve((long long)D * H * 2);   // [H,D]
    __bf16* W2T = (__bf16*)carve((long long)H * D * 2);   // [D,H]
    __bf16* Qb  = (__bf16*)carve(nBND * 2);
    __bf16* Kb  = (__bf16*)carve(nBND * 2);
    __bf16* Vb  = (__bf16*)carve(nBND * 2);
    __bf16* Vt  = (__bf16*)carve(nBND * 2);               // per-batch [D,N]
    float*  Sf  = (float*) carve(nS * 4);
    __bf16* Pb  = (__bf16*)carve(nS * 2);
    __bf16* Cx  = (__bf16*)carve(nBND * 2);
    float*  Yf  = (float*) carve(nBND * 4);
    float*  O1f = (float*) carve(nBND * 4);
    __bf16* O1b = (__bf16*)carve(nBND * 2);
    float*  Hf  = (float*) carve(nBNH * 4);
    __bf16* Hb  = (__bf16*)carve(nBNH * 2);
    float*  Ff  = (float*) carve(nBND * 4);

    const dim3 blk(256);
    auto cgrid = [](long long n) { return dim3((unsigned)((n / 4 + 255) / 256)); };

    // ---- 1) convert activations; transpose-convert weights ----
    cvt_f32_bf16<<<cgrid(nBND), blk, 0, stream>>>(q_in, Xq, nBND / 4);
    cvt_f32_bf16<<<cgrid(nBND), blk, 0, stream>>>(k_in, Xk, nBND / 4);
    {
        dim3 gdd(D / 32, D / 32, 1);
        transpose_to_bf16<float><<<gdd, blk, 0, stream>>>(Wq, WqT, D, D, 0, 0);
        transpose_to_bf16<float><<<gdd, blk, 0, stream>>>(Wk, WkT, D, D, 0, 0);
        transpose_to_bf16<float><<<gdd, blk, 0, stream>>>(Wv, WvT, D, D, 0, 0);
        transpose_to_bf16<float><<<gdd, blk, 0, stream>>>(Wo, WoT, D, D, 0, 0);
        transpose_to_bf16<float><<<dim3(H / 32, D / 32, 1), blk, 0, stream>>>(W1, W1T, D, H, 0, 0);
        transpose_to_bf16<float><<<dim3(D / 32, H / 32, 1), blk, 0, stream>>>(W2, W2T, H, D, 0, 0);
    }

    // ---- 2) Q/K/V projections: [32768,768] x [768,768]^T ----
    {
        dim3 g(D / TBN, M / TBM, 1);
        gemm_nt_wmma<true><<<g, blk, 0, stream>>>(Xq, WqT, bq, nullptr, Qb, M, D, D, 1.0f, 0, 0, 0);
        gemm_nt_wmma<true><<<g, blk, 0, stream>>>(Xk, WkT, bk, nullptr, Kb, M, D, D, 1.0f, 0, 0, 0);
        gemm_nt_wmma<true><<<g, blk, 0, stream>>>(Xk, WvT, bv, nullptr, Vb, M, D, D, 1.0f, 0, 0, 0);
    }

    // ---- 3) V^T per batch: [N,D] -> [D,N] ----
    transpose_to_bf16<__bf16><<<dim3(D / 32, N / 32, B), blk, 0, stream>>>(
        Vb, Vt, N, D, (long long)N * D, (long long)N * D);

    // ---- 4) scores = Q @ K^T / sqrt(D): batched NT GEMM ----
    {
        dim3 g(N / TBN, N / TBM, B);
        const float alpha = 0.03608439182435161f;  // 1/sqrt(768)
        gemm_nt_wmma<false><<<g, blk, 0, stream>>>(
            Qb, Kb, nullptr, Sf, nullptr, N, N, D, alpha,
            (long long)N * D, (long long)N * D, (long long)N * N);
    }

    // ---- 5) softmax rows -> bf16 probabilities ----
    softmax_rows_2048<<<dim3(M), blk, 0, stream>>>(Sf, Pb);

    // ---- 6) context = P @ V = P @ (V^T)^T: batched NT GEMM ----
    {
        dim3 g(D / TBN, N / TBM, B);
        gemm_nt_wmma<true><<<g, blk, 0, stream>>>(
            Pb, Vt, nullptr, nullptr, Cx, N, D, N, 1.0f,
            (long long)N * N, (long long)N * D, (long long)N * D);
    }

    // ---- 7) output projection + residual + LN1 ----
    {
        dim3 g(D / TBN, M / TBM, 1);
        gemm_nt_wmma<false><<<g, blk, 0, stream>>>(Cx, WoT, bo, Yf, nullptr, M, D, D, 1.0f, 0, 0, 0);
    }
    add_layernorm<D><<<dim3(M), blk, 0, stream>>>(q_in, Yf, g1, b1, O1f, O1b);

    // ---- 8) FFN: Linear -> LN -> GELU -> Linear ----
    {
        dim3 g(H / TBN, M / TBM, 1);
        gemm_nt_wmma<false><<<g, blk, 0, stream>>>(O1b, W1T, bf1, Hf, nullptr, M, H, D, 1.0f, 0, 0, 0);
    }
    layernorm_gelu<H><<<dim3(M), blk, 0, stream>>>(Hf, gf, bff, Hb);
    {
        dim3 g(D / TBN, M / TBM, 1);
        gemm_nt_wmma<false><<<g, blk, 0, stream>>>(Hb, W2T, bf2, Ff, nullptr, M, D, H, 1.0f, 0, 0, 0);
    }

    // ---- 9) final residual + LN2 -> d_out (f32) ----
    add_layernorm<D><<<dim3(M), blk, 0, stream>>>(O1f, Ff, g2, b2, (float*)d_out, nullptr);

    (void)in_sizes; (void)n_in; (void)out_size; (void)ws_size;
}
